// DifferentialAttention_85667417686300
// MI455X (gfx1250) — matrix operand order repaired and checked by reading, not compiled, not run on hardware
//
#include <hip/hip_runtime.h>
#include <hip/hip_bf16.h>

// ---------------------------------------------------------------------------
// Differential attention for MI455X (gfx1250), wave32 + WMMA bf16.
//   B=1, N=2048, C=1024, H=16, hd=64, rot_dim=128, DEPTH=0 -> lambda_init=0.2
// GEMMs are NT (K contiguous on both operands). GEMM B-tiles are staged in LDS
// shared by 8 waves, double buffered with async global->LDS copies
// (global_load_async_to_lds_b128 / s_wait_asynccnt via inline asm).
// ---------------------------------------------------------------------------

typedef __attribute__((ext_vector_type(16))) __bf16 v16bf;
typedef __attribute__((ext_vector_type(8)))  __bf16 v8bf;
typedef __attribute__((ext_vector_type(8)))  float  v8f;

#define N_SEQ  2048
#define CDIM   1024
#define HEADS  16
#define HD     64
#define QKVD   6144   // 6*C
#define AOD    2048   // 2*C

// ---- CDNA5 async global->LDS copy (ASYNCcnt-tracked) -----------------------

__device__ __forceinline__ void async_cp16(const __bf16* g, __bf16* l) {
#if defined(__gfx1250__)
  // LDS byte address = low 32 bits of the generic pointer (ISA 10.2 aperture)
  unsigned int loff = (unsigned int)(unsigned long long)l;
  asm volatile("global_load_async_to_lds_b128 %0, %1, off"
               :: "v"(loff), "v"(g)
               : "memory");
#else
  *(v8bf*)l = *(const v8bf*)g;
#endif
}

template <int N>
__device__ __forceinline__ void wait_asynccnt() {
#if defined(__gfx1250__)
  if constexpr (N == 0) asm volatile("s_wait_asynccnt 0" ::: "memory");
  else                  asm volatile("s_wait_asynccnt 1" ::: "memory");
#endif
}

// ---- WMMA helpers ----------------------------------------------------------

__device__ __forceinline__ v16bf cat8(v8bf lo, v8bf hi) {
  return __builtin_shufflevector(lo, hi, 0,1,2,3,4,5,6,7,8,9,10,11,12,13,14,15);
}

// A-matrix 16x32 bf16 fragment. rowptr = &A[row_of_this_lane][0].
// lanes 0-15 hold K {k..k+7, k+16..k+23}; lanes 16-31 hold K {k+8..k+15, k+24..k+31}.
__device__ __forceinline__ v16bf load_a_frag(const __bf16* rowptr, int k, int hw) {
  const v8bf* p0 = (const v8bf*)(rowptr + k + hw * 8);
  const v8bf* p1 = (const v8bf*)(rowptr + k + 16 + hw * 8);
  return cat8(*p0, *p1);
}

// B-matrix 32x16 bf16 fragment. rowptr = &B[col_of_this_lane][0] (NT operand).
// lanes 0-15 hold K {k..k+7, k+16..k+23}; lanes 16-31 hold K {k+8..k+15, k+24..k+31}.
__device__ __forceinline__ v16bf load_b_frag(const __bf16* rowptr, int k, int hw) {
  const v8bf* p0 = (const v8bf*)(rowptr + k + hw * 8);
  const v8bf* p1 = (const v8bf*)(rowptr + k + 16 + hw * 8);
  return cat8(*p0, *p1);
}

__device__ __forceinline__ v8f wmma_bf16(v16bf a, v16bf b, v8f c) {
  // (neg_a, A, neg_b, B, c_mod, C, reuse_a, reuse_b)
  return __builtin_amdgcn_wmma_f32_16x16x32_bf16(false, a, false, b, (short)0, c,
                                                 false, false);
}

// ---- fp32 -> bf16 cast -----------------------------------------------------

__global__ void cast_f32_bf16(const float* __restrict__ src,
                              __bf16* __restrict__ dst, int n) {
  int i = blockIdx.x * blockDim.x + threadIdx.x;
  if (i < n) dst[i] = (__bf16)src[i];
}

// ---- Block-cooperative NT GEMM: C[M,D] = A[M,K] * B[D,K]^T ----------------
// Block = 256 threads = 8 waves, tile 128(M) x 64(D).
// B tile (64 rows x 32 K = 4 KB) staged in LDS once per k-step, shared by all
// 8 waves; double buffered with async global->LDS copies. Each wave owns 16 M
// rows and 4 fp32 accumulators (16x64).

template <typename OutT>
__global__ void gemm_nt(const __bf16* __restrict__ A, const __bf16* __restrict__ B,
                        OutT* __restrict__ C, int M, int D, int K) {
  __shared__ __bf16 bbuf[2][64 * 32];  // 2 x 4 KB
  const int tid  = threadIdx.x;
  const int lane = tid & 31;
  const int wave = tid >> 5;
  const int hw = lane >> 4, l16 = lane & 15;
  const int db = blockIdx.x * 64;
  const int mb = blockIdx.y * 128 + wave * 16;

  const __bf16* arow = A + (size_t)(mb + l16) * K;

  // this thread's 16B chunk of the B tile: row r (of 64), 8-elem chunk c (of 4)
  const int r = tid >> 2, c = (tid & 3) * 8;
  const __bf16* bsrc = B + (size_t)(db + r) * K + c;
  __bf16* const ldst0 = &bbuf[0][r * 32 + c];
  __bf16* const ldst1 = &bbuf[1][r * 32 + c];

  v8f acc0 = {}, acc1 = {}, acc2 = {}, acc3 = {};
  const int nsteps = K >> 5;

  async_cp16(bsrc, ldst0);
  for (int i = 0; i < nsteps; ++i) {
    if (i + 1 < nsteps) {
      async_cp16(bsrc + (i + 1) * 32, ((i + 1) & 1) ? ldst1 : ldst0);
      wait_asynccnt<1>();  // current tile's copy (issued earlier, in order) done
    } else {
      wait_asynccnt<0>();
    }
    __syncthreads();

    const __bf16* cur = bbuf[i & 1];
    v16bf af = load_a_frag(arow, i * 32, hw);
    v16bf b0 = load_b_frag(cur + (size_t)(0 * 16 + l16) * 32, 0, hw);
    v16bf b1 = load_b_frag(cur + (size_t)(1 * 16 + l16) * 32, 0, hw);
    v16bf b2 = load_b_frag(cur + (size_t)(2 * 16 + l16) * 32, 0, hw);
    v16bf b3 = load_b_frag(cur + (size_t)(3 * 16 + l16) * 32, 0, hw);
    acc0 = wmma_bf16(af, b0, acc0);
    acc1 = wmma_bf16(af, b1, acc1);
    acc2 = wmma_bf16(af, b2, acc2);
    acc3 = wmma_bf16(af, b3, acc3);
    __syncthreads();
  }

  // C/D layout: VGPR vi -> row (hw*8 + vi), lane l16 -> col
  const int row = mb + hw * 8;
#pragma unroll
  for (int vi = 0; vi < 8; ++vi) {
    size_t rr = (size_t)(row + vi) * D + db + l16;
    C[rr + 0]  = (OutT)acc0[vi];
    C[rr + 16] = (OutT)acc1[vi];
    C[rr + 32] = (OutT)acc2[vi];
    C[rr + 48] = (OutT)acc3[vi];
  }
}

// ---- lambda = exp(lq1.lk1) - exp(lq2.lk2) + 0.2 (one wave) ----------------

__global__ void lambda_kernel(const float* __restrict__ lq1, const float* __restrict__ lq2,
                              const float* __restrict__ lk1, const float* __restrict__ lk2,
                              float* __restrict__ lam) {
  int lane = threadIdx.x;
  float d1 = lq1[lane] * lk1[lane] + lq1[lane + 32] * lk1[lane + 32];
  float d2 = lq2[lane] * lk2[lane] + lq2[lane + 32] * lk2[lane + 32];
  for (int m = 16; m >= 1; m >>= 1) {
    d1 += __shfl_xor(d1, m, 32);
    d2 += __shfl_xor(d2, m, 32);
  }
  if (lane == 0) lam[0] = __expf(d1) - __expf(d2) + 0.2f;  // lambda_init(DEPTH=0)
}

// ---- RoPE + repack qkv[bf16 N x 6C] -> bf16 q1,q2,k1,k2 [H][N][64], vT [H][128][N]

__global__ void rope_pack(const __bf16* __restrict__ qkv,
                          __bf16* __restrict__ q1o, __bf16* __restrict__ q2o,
                          __bf16* __restrict__ k1o, __bf16* __restrict__ k2o,
                          __bf16* __restrict__ vT) {
  int t = blockIdx.x * blockDim.x + threadIdx.x;  // N*H*64 = 2M threads
  int e = t & 63;
  int h = (t >> 6) & 15;
  int n = t >> 10;
  const __bf16* row = qkv + (size_t)n * QKVD + h * HD + e;
  float q1v = (float)row[0],        q2v = (float)row[1 * CDIM];
  float k1v = (float)row[2 * CDIM], k2v = (float)row[3 * CDIM];
  float v1  = (float)row[4 * CDIM], v2  = (float)row[5 * CDIM];

  // inv_freq = 10000^(-e/64); emb repeats, so cos/sin identical for e and e+64
  float inv = __powf(10000.f, -(float)e * (1.f / 64.f));
  float th = (float)n * inv;
  float s, cs;
  __sincosf(th, &s, &cs);

  size_t hn = ((size_t)h * N_SEQ + n) * HD + e;
  q1o[hn] = (__bf16)(q1v * cs - q2v * s);
  q2o[hn] = (__bf16)(q2v * cs + q1v * s);
  k1o[hn] = (__bf16)(k1v * cs - k2v * s);
  k2o[hn] = (__bf16)(k2v * cs + k1v * s);

  size_t vb = ((size_t)h * 128 + e) * N_SEQ + n;  // transposed: [h][e][n]
  vT[vb]              = (__bf16)v1;
  vT[vb + 64 * N_SEQ] = (__bf16)v2;
}

// ---- Pass A: per-row softmax stats (m, l) for both score streams ----------
// 4 independent waves per block; one wave per (head, 16-row block).

__global__ void attn_stats(const __bf16* __restrict__ q1, const __bf16* __restrict__ q2,
                           const __bf16* __restrict__ k1, const __bf16* __restrict__ k2,
                           float* __restrict__ m1s, float* __restrict__ l1s,
                           float* __restrict__ m2s, float* __restrict__ l2s) {
  const int lane = threadIdx.x & 31;
  const int hw = lane >> 4, l16 = lane & 15;
  const int gid = blockIdx.x * 4 + (threadIdx.x >> 5);
  const int h  = gid >> 7;
  const int ib = gid & 127;

  const __bf16* qb1 = q1 + ((size_t)h * N_SEQ + ib * 16 + l16) * HD;
  const __bf16* qb2 = q2 + ((size_t)h * N_SEQ + ib * 16 + l16) * HD;
  v16bf qa10 = load_a_frag(qb1, 0, hw), qa11 = load_a_frag(qb1, 32, hw);
  v16bf qa20 = load_a_frag(qb2, 0, hw), qa21 = load_a_frag(qb2, 32, hw);

  const __bf16* kh1 = k1 + (size_t)h * N_SEQ * HD;
  const __bf16* kh2 = k2 + (size_t)h * N_SEQ * HD;

  float m1[8], l1[8], m2[8], l2[8];
#pragma unroll
  for (int vi = 0; vi < 8; ++vi) { m1[vi] = m2[vi] = -1e30f; l1[vi] = l2[vi] = 0.f; }

  for (int jb = 0; jb < 128; ++jb) {
    const __bf16* kr1 = kh1 + (size_t)(jb * 16 + l16) * HD;
    const __bf16* kr2 = kh2 + (size_t)(jb * 16 + l16) * HD;
    v8f s1 = {}, s2 = {};
    s1 = wmma_bf16(qa10, load_b_frag(kr1, 0, hw),  s1);
    s1 = wmma_bf16(qa11, load_b_frag(kr1, 32, hw), s1);
    s2 = wmma_bf16(qa20, load_b_frag(kr2, 0, hw),  s2);
    s2 = wmma_bf16(qa21, load_b_frag(kr2, 32, hw), s2);
#pragma unroll
    for (int vi = 0; vi < 8; ++vi) {
      float v1 = s1[vi] * 0.125f, v2 = s2[vi] * 0.125f;  // scale = hd^-0.5
      float t1 = v1, t2 = v2;
      for (int m = 8; m >= 1; m >>= 1) {                 // max over 16 cols
        t1 = fmaxf(t1, __shfl_xor(t1, m, 32));
        t2 = fmaxf(t2, __shfl_xor(t2, m, 32));
      }
      float nm1 = fmaxf(m1[vi], t1), nm2 = fmaxf(m2[vi], t2);
      float e1 = __expf(v1 - nm1), e2 = __expf(v2 - nm2);
      for (int m = 8; m >= 1; m >>= 1) {                 // sum over 16 cols
        e1 += __shfl_xor(e1, m, 32);
        e2 += __shfl_xor(e2, m, 32);
      }
      l1[vi] = l1[vi] * __expf(m1[vi] - nm1) + e1;
      l2[vi] = l2[vi] * __expf(m2[vi] - nm2) + e2;
      m1[vi] = nm1; m2[vi] = nm2;
    }
  }
  if (l16 == 0) {
#pragma unroll
    for (int vi = 0; vi < 8; ++vi) {
      size_t idx = (size_t)h * N_SEQ + ib * 16 + hw * 8 + vi;
      m1s[idx] = m1[vi]; l1s[idx] = l1[vi];
      m2s[idx] = m2[vi]; l2s[idx] = l2[vi];
    }
  }
}

// ---- Pass B: p = softmax1 - lam*softmax2 ; out = p @ V ---------------------
// 4 waves/block, each with its own LDS slice for C-layout -> A-layout P tiles.

__global__ void attn_out(const __bf16* __restrict__ q1, const __bf16* __restrict__ q2,
                         const __bf16* __restrict__ k1, const __bf16* __restrict__ k2,
                         const __bf16* __restrict__ vT,
                         const float* __restrict__ m1s, const float* __restrict__ l1s,
                         const float* __restrict__ m2s, const float* __restrict__ l2s,
                         const float* __restrict__ lamp,
                         __bf16* __restrict__ ao) {
  __shared__ __bf16 plds[4][16 * 32];
  const int lane = threadIdx.x & 31;
  const int wv = threadIdx.x >> 5;
  const int hw = lane >> 4, l16 = lane & 15;
  const int gid = blockIdx.x * 4 + wv;
  const int h  = gid >> 7;
  const int ib = gid & 127;
  const float lam = lamp[0];
  __bf16* lds = plds[wv];

  const __bf16* qb1 = q1 + ((size_t)h * N_SEQ + ib * 16 + l16) * HD;
  const __bf16* qb2 = q2 + ((size_t)h * N_SEQ + ib * 16 + l16) * HD;
  v16bf qa10 = load_a_frag(qb1, 0, hw), qa11 = load_a_frag(qb1, 32, hw);
  v16bf qa20 = load_a_frag(qb2, 0, hw), qa21 = load_a_frag(qb2, 32, hw);

  const __bf16* kh1 = k1 + (size_t)h * N_SEQ * HD;
  const __bf16* kh2 = k2 + (size_t)h * N_SEQ * HD;
  const __bf16* vh  = vT + (size_t)h * 128 * N_SEQ;

  float m1[8], il1[8], m2[8], il2[8];
#pragma unroll
  for (int vi = 0; vi < 8; ++vi) {
    size_t idx = (size_t)h * N_SEQ + ib * 16 + hw * 8 + vi;
    m1[vi] = m1s[idx]; il1[vi] = 1.f / l1s[idx];
    m2[vi] = m2s[idx]; il2[vi] = 1.f / l2s[idx];
  }

  v8f acc[8] = {};
  for (int j0 = 0; j0 < N_SEQ; j0 += 32) {
#pragma unroll
    for (int jt = 0; jt < 2; ++jt) {
      const __bf16* kr1 = kh1 + (size_t)(j0 + jt * 16 + l16) * HD;
      const __bf16* kr2 = kh2 + (size_t)(j0 + jt * 16 + l16) * HD;
      v8f s1 = {}, s2 = {};
      s1 = wmma_bf16(qa10, load_b_frag(kr1, 0, hw),  s1);
      s1 = wmma_bf16(qa11, load_b_frag(kr1, 32, hw), s1);
      s2 = wmma_bf16(qa20, load_b_frag(kr2, 0, hw),  s2);
      s2 = wmma_bf16(qa21, load_b_frag(kr2, 32, hw), s2);
#pragma unroll
      for (int vi = 0; vi < 8; ++vi) {
        float p = __expf(s1[vi] * 0.125f - m1[vi]) * il1[vi]
                - lam * __expf(s2[vi] * 0.125f - m2[vi]) * il2[vi];
        lds[(hw * 8 + vi) * 32 + jt * 16 + l16] = (__bf16)p;
      }
    }
    __syncthreads();
    v16bf pf = load_a_frag(&lds[l16 * 32], 0, hw);  // 16x32 P tile as A operand
#pragma unroll
    for (int et = 0; et < 8; ++et) {
      const __bf16* vrow = vh + (size_t)(et * 16 + l16) * N_SEQ;
      acc[et] = wmma_bf16(pf, load_b_frag(vrow, j0, hw), acc[et]);
    }
    __syncthreads();
  }
  // store attn output [n][2C] row-major (bf16) for the final projection GEMM
#pragma unroll
  for (int et = 0; et < 8; ++et)
#pragma unroll
    for (int vi = 0; vi < 8; ++vi)
      ao[(size_t)(ib * 16 + hw * 8 + vi) * AOD + h * 128 + et * 16 + l16] =
          (__bf16)acc[et][vi];
}

// ---------------------------------------------------------------------------

extern "C" void kernel_launch(void* const* d_in, const int* in_sizes, int n_in,
                              void* d_out, int out_size, void* d_ws, size_t ws_size,
                              hipStream_t stream) {
  const float* x      = (const float*)d_in[0];
  const float* W_qkv  = (const float*)d_in[1];
  const float* W_proj = (const float*)d_in[2];
  const float* lq1    = (const float*)d_in[3];
  const float* lq2    = (const float*)d_in[4];
  const float* lk1    = (const float*)d_in[5];
  const float* lk2    = (const float*)d_in[6];
  float* out = (float*)d_out;

  // workspace carve (256B aligned), ~77 MB total
  char* p = (char*)d_ws;
  auto alloc = [&](size_t bytes) -> void* {
    void* r = (void*)p;
    p += (bytes + 255) & ~(size_t)255;
    return r;
  };
  __bf16* xbf  = (__bf16*)alloc((size_t)N_SEQ * CDIM * 2);        // 4 MB
  __bf16* wqbf = (__bf16*)alloc((size_t)QKVD * CDIM * 2);         // 12 MB
  __bf16* wpbf = (__bf16*)alloc((size_t)CDIM * AOD * 2);          // 4 MB
  __bf16* qkvb = (__bf16*)alloc((size_t)N_SEQ * QKVD * 2);        // 24 MB
  __bf16* q1b  = (__bf16*)alloc((size_t)HEADS * N_SEQ * HD * 2);  // 4 MB
  __bf16* q2b  = (__bf16*)alloc((size_t)HEADS * N_SEQ * HD * 2);
  __bf16* k1b  = (__bf16*)alloc((size_t)HEADS * N_SEQ * HD * 2);
  __bf16* k2b  = (__bf16*)alloc((size_t)HEADS * N_SEQ * HD * 2);
  __bf16* vTb  = (__bf16*)alloc((size_t)HEADS * 128 * N_SEQ * 2); // 8 MB
  __bf16* aob  = (__bf16*)alloc((size_t)N_SEQ * AOD * 2);         // 8 MB
  float*  m1s  = (float*)alloc((size_t)HEADS * N_SEQ * 4);
  float*  l1s  = (float*)alloc((size_t)HEADS * N_SEQ * 4);
  float*  m2s  = (float*)alloc((size_t)HEADS * N_SEQ * 4);
  float*  l2s  = (float*)alloc((size_t)HEADS * N_SEQ * 4);
  float*  lam  = (float*)alloc(256);

  // 1) casts to bf16
  cast_f32_bf16<<<(N_SEQ * CDIM) / 256, 256, 0, stream>>>(x, xbf, N_SEQ * CDIM);
  cast_f32_bf16<<<(QKVD * CDIM) / 256, 256, 0, stream>>>(W_qkv, wqbf, QKVD * CDIM);
  cast_f32_bf16<<<(CDIM * AOD) / 256, 256, 0, stream>>>(W_proj, wpbf, CDIM * AOD);

  // 2) QKV GEMM: qkv[N,6C] = x[N,C] @ W_qkv[6C,C]^T  (bf16 out)
  gemm_nt<__bf16><<<dim3(QKVD / 64, N_SEQ / 128), 256, 0, stream>>>(
      xbf, wqbf, qkvb, N_SEQ, QKVD, CDIM);

  // 3) lambda scalar
  lambda_kernel<<<1, 32, 0, stream>>>(lq1, lq2, lk1, lk2, lam);

  // 4) RoPE + repack (q1,q2,k1,k2 per-head row-major; V transposed)
  rope_pack<<<(N_SEQ * HEADS * HD) / 256, 256, 0, stream>>>(
      qkvb, q1b, q2b, k1b, k2b, vTb);

  // 5) softmax stats for both score streams (one wave per 16 query rows/head)
  attn_stats<<<HEADS * (N_SEQ / 16) / 4, 128, 0, stream>>>(
      q1b, q2b, k1b, k2b, m1s, l1s, m2s, l2s);

  // 6) differential probs + P@V
  attn_out<<<HEADS * (N_SEQ / 16) / 4, 128, 0, stream>>>(
      q1b, q2b, k1b, k2b, vTb, m1s, l1s, m2s, l2s, lam, aob);

  // 7) output projection: out[N,C] = ao[N,2C] @ W_proj[C,2C]^T  (f32 out)
  gemm_nt<float><<<dim3(CDIM / 64, N_SEQ / 128), 256, 0, stream>>>(
      aob, wpbf, out, N_SEQ, CDIM, AOD);
}
